// Encoder_67121748901904
// MI455X (gfx1250) — matrix operand order repaired and checked by reading, not compiled, not run on hardware
//
#include <hip/hip_runtime.h>
#include <hip/hip_bf16.h>
#include <cstdint>
#include <cstddef>

typedef __bf16 bf16_t;
typedef __attribute__((ext_vector_type(16))) __bf16 v16bf;
typedef __attribute__((ext_vector_type(8)))  float  v8f;

static __host__ int div_up(int a, int b) { return (a + b - 1) / b; }

// Native f32 atomic add (global_atomic_add_f32), no CAS fallback.
__device__ inline void atomic_add_f32(float* p, float v) {
  unsafeAtomicAdd(p, v);
}

// ---------------- degree / normalization ----------------
__global__ void k_dinv_init(float* __restrict__ dinv, int n) {
  int i = blockIdx.x * blockDim.x + threadIdx.x;
  if (i < n) dinv[i] = 1.0f;  // self-loop counts as degree 1
}

__global__ void k_deg_accum(const int* __restrict__ dst, float* __restrict__ dinv, int e) {
  int i = blockIdx.x * blockDim.x + threadIdx.x;
  if (i < e) atomic_add_f32(&dinv[dst[i]], 1.0f);
}

__global__ void k_dinv_finish(float* __restrict__ dinv, int n) {
  int i = blockIdx.x * blockDim.x + threadIdx.x;
  if (i < n) dinv[i] = rsqrtf(dinv[i]);
}

// ---------------- f32 -> bf16 conversion (activations) ----------------
__global__ void k_f32_to_bf16(const float* __restrict__ x, bf16_t* __restrict__ y, int n) {
  int i = blockIdx.x * blockDim.x + threadIdx.x;
  if (i < n) y[i] = (bf16_t)x[i];
}

// ---------------- weight repack: f32 [FIN,FOUT] -> bf16 WMMA fragment order --
// Fragment order: for kstep ks (K block of 32), ntile t, lane l (0..31):
//   Wf[((ks*NT + t)*32 + l)*16 + j]  holds  B[k][n]  with
//   n = t*16 + (l&15), half = l>>4, k = ks*32 + half*8 + (j&7) + (j>>3)*16
// so the GEMM reads one aligned 32B v16bf per fragment per lane.
template <int FIN, int FOUT>
__global__ void k_pack_w(const float* __restrict__ W, bf16_t* __restrict__ Wf) {
  constexpr int NT = FOUT / 16;
  int tid = blockIdx.x * blockDim.x + threadIdx.x;
  if (tid >= FIN * FOUT) return;
  const int j    = tid & 15;
  const int lane = (tid >> 4) & 31;
  const int frag = tid >> 9;            // ks*NT + t
  const int t    = frag % NT;
  const int ks   = frag / NT;
  const int half = lane >> 4;
  const int n    = (t << 4) + (lane & 15);
  const int k    = ks * 32 + half * 8 + (j & 7) + ((j >> 3) << 4);
  Wf[tid] = (bf16_t)W[(size_t)k * FOUT + n];
}

// ---------------- WMMA GEMM + dinv row-scale epilogue ----------------
// Computes G[i,:] = (Xb @ W)[i,:] * dinv[i]  and  ACC = G (self-loop init).
// One wave per 16-node row tile; FOUT/16 accumulator tiles held in VGPRs.
template <int FIN, int FOUT>
__global__ void k_gcn_gemm(const bf16_t* __restrict__ Xb,   // [N, FIN] row-major
                           const bf16_t* __restrict__ Wf,   // fragment-packed weights
                           const float*  __restrict__ dinv,
                           float* __restrict__ G,
                           float* __restrict__ ACC,
                           int mtiles) {
  const int wave = threadIdx.x >> 5;
  const int lane = threadIdx.x & 31;
  const int mt   = blockIdx.x * (blockDim.x >> 5) + wave;
  if (mt >= mtiles) return;                 // wave-uniform: EXEC stays all-ones for WMMA
  const int l15  = lane & 15;
  const int half = lane >> 4;
  constexpr int NT = FOUT / 16;

  v8f zero = {};
  v8f acc[NT];
#pragma unroll
  for (int t = 0; t < NT; ++t) acc[t] = zero;

  const int m = (mt << 4) + l15;            // A-matrix row this lane supplies
  const v16bf* __restrict__ Bfrag = (const v16bf*)Wf;

  union Frag { uint4 u[2]; v16bf v; };

#pragma unroll
  for (int k0 = 0; k0 < FIN; k0 += 32) {
    const int ks = k0 >> 5;
    // A fragment 16x32 bf16 (ISA 7.12.2): lane needs K = half*8 + {0..7} and
    // 16 + half*8 + {0..7} -> two contiguous 16B runs -> two b128 loads.
    Frag af;
    const bf16_t* arow = Xb + (size_t)m * FIN + k0 + half * 8;
    af.u[0] = *(const uint4*)(arow);
    af.u[1] = *(const uint4*)(arow + 16);
#pragma unroll
    for (int t = 0; t < NT; ++t) {
      // B fragment: one aligned 32B load from the pre-packed weight buffer.
      v16bf b = Bfrag[(size_t)(ks * NT + t) * 32 + lane];
      acc[t] = __builtin_amdgcn_wmma_f32_16x16x32_bf16(
          /*neg_a=*/false, af.v, /*neg_b=*/false, b,
          /*c_mod=*/(short)0, acc[t], /*reuse_a=*/false, /*reuse_b=*/false);
    }
  }

  // Epilogue: C/D layout — VGPR r holds row M = r + 8*half, column = lane&15.
#pragma unroll
  for (int t = 0; t < NT; ++t) {
#pragma unroll
    for (int r = 0; r < 8; ++r) {
      const int row = (mt << 4) + r + half * 8;
      const float g = acc[t][r] * dinv[row];
      const size_t idx = (size_t)row * FOUT + (t << 4) + l15;
      G[idx]   = g;   // gather source
      ACC[idx] = g;   // self-loop contribution pre-seeded
    }
  }
}

// ---------------- edge scatter: ACC[dst] += G[src] ----------------
// thread handles 4 consecutive features of one edge; chunks = F/4 (power of 2)
__global__ void k_scatter(const int* __restrict__ src, const int* __restrict__ dst,
                          const float* __restrict__ G, float* __restrict__ ACC,
                          int E, int F, int lgC) {
  int tid = blockIdx.x * blockDim.x + threadIdx.x;
  int e = tid >> lgC;
  if (e >= E) return;
  int c = tid & ((1 << lgC) - 1);
  int s = src[e], d = dst[e];
  const float4 g = *(const float4*)(G + (size_t)s * F + c * 4);
  float* out = ACC + (size_t)d * F + c * 4;
  atomic_add_f32(out + 0, g.x);
  atomic_add_f32(out + 1, g.y);
  atomic_add_f32(out + 2, g.z);
  atomic_add_f32(out + 3, g.w);
}

// ---------------- out = relu(dinv*ACC + b) -> bf16 next-layer input ----------
__global__ void k_post_relu_bf16(const float* __restrict__ ACC, const float* __restrict__ dinv,
                                 const float* __restrict__ bias, bf16_t* __restrict__ Xb,
                                 int total, int lgF, int F) {
  int tid = blockIdx.x * blockDim.x + threadIdx.x;
  if (tid >= total) return;
  int i = tid >> lgF;
  int f = tid & (F - 1);
  float v = dinv[i] * ACC[tid] + bias[f];
  Xb[tid] = (bf16_t)fmaxf(v, 0.0f);
}

// ---------------- final layer: H = dinv*ACC + b (f32, no relu) --------------
__global__ void k_post_f32(const float* __restrict__ ACC, const float* __restrict__ dinv,
                           const float* __restrict__ bias, float* __restrict__ H,
                           int total, int lgF, int F) {
  int tid = blockIdx.x * blockDim.x + threadIdx.x;
  if (tid >= total) return;
  int i = tid >> lgF;
  int f = tid & (F - 1);
  H[tid] = dinv[i] * ACC[tid] + bias[f];
}

// ---------------- global max pool (batch sorted -> binary search range) -----
__global__ void k_maxpool(const float* __restrict__ H, const int* __restrict__ batch,
                          float* __restrict__ out, int N, int F) {
  const int g = blockIdx.x;
  const int f = threadIdx.x;  // blockDim.x == F == 64
  // lower_bound(g)
  int lo = 0, hi = N;
  while (lo < hi) { int mid = (lo + hi) >> 1; if (batch[mid] < g) lo = mid + 1; else hi = mid; }
  const int s = lo;
  // lower_bound(g+1)
  hi = N;
  while (lo < hi) { int mid = (lo + hi) >> 1; if (batch[mid] < g + 1) lo = mid + 1; else hi = mid; }
  const int e = lo;
  float m = -INFINITY;
  for (int i = s; i < e; ++i) m = fmaxf(m, H[(size_t)i * F + f]);
  out[g * F + f] = m;
}

// ---------------- orchestration ----------------
extern "C" void kernel_launch(void* const* d_in, const int* in_sizes, int n_in,
                              void* d_out, int out_size, void* d_ws, size_t ws_size,
                              hipStream_t stream) {
  const float* data  = (const float*)d_in[0];   // [N,128]
  const int*   eidx  = (const int*)d_in[1];     // [2,E]
  const int*   batch = (const int*)d_in[2];     // [N]
  const float* W1 = (const float*)d_in[3];
  const float* b1 = (const float*)d_in[4];
  const float* W2 = (const float*)d_in[5];
  const float* b2 = (const float*)d_in[6];
  const float* W3 = (const float*)d_in[7];
  const float* b3 = (const float*)d_in[8];

  const int N = in_sizes[2];          // 50000
  const int E = in_sizes[1] / 2;      // 600000
  const int IN_DIM = 128, HID = 64;
  const int* src = eidx;
  const int* dst = eidx + E;

  // workspace carve-up (256B aligned)
  char* ws = (char*)d_ws;
  auto align256 = [](size_t x) { return (x + 255) & ~(size_t)255; };
  size_t off = 0;
  float*  dinv = (float*)(ws + off);  off += align256((size_t)N * 4);
  bf16_t* Xb   = (bf16_t*)(ws + off); off += align256((size_t)N * 2 * HID * 2);       // N x 128 bf16
  bf16_t* Wf   = (bf16_t*)(ws + off); off += align256((size_t)2 * HID * 2 * HID * 2); // 128x128 bf16
  float*  G    = (float*)(ws + off);  off += align256((size_t)N * 2 * HID * 4);       // N x 128 f32
  float*  ACC  = (float*)(ws + off);  off += align256((size_t)N * 2 * HID * 4);       // N x 128 f32
  (void)ws_size;

  const int TPB = 256;
  const int mtiles = div_up(N, 16);
  const int gemm_waves_per_blk = 4;        // blockDim = 128 -> 4 m-tiles/block
  const int gemm_grid = div_up(mtiles, gemm_waves_per_blk);

  // degree -> dinv = rsqrt(1 + in_degree)
  k_dinv_init<<<div_up(N, TPB), TPB, 0, stream>>>(dinv, N);
  k_deg_accum<<<div_up(E, TPB), TPB, 0, stream>>>(dst, dinv, E);
  k_dinv_finish<<<div_up(N, TPB), TPB, 0, stream>>>(dinv, N);

  // ---- layer 1: 128 -> 64, relu ----
  k_f32_to_bf16<<<div_up(N * IN_DIM, TPB), TPB, 0, stream>>>(data, Xb, N * IN_DIM);
  k_pack_w<128, 64><<<div_up(IN_DIM * HID, TPB), TPB, 0, stream>>>(W1, Wf);
  k_gcn_gemm<128, 64><<<gemm_grid, 32 * gemm_waves_per_blk, 0, stream>>>(Xb, Wf, dinv, G, ACC, mtiles);
  k_scatter<<<div_up(E * (64 / 4), TPB), TPB, 0, stream>>>(src, dst, G, ACC, E, 64, 4);
  k_post_relu_bf16<<<div_up(N * 64, TPB), TPB, 0, stream>>>(ACC, dinv, b1, Xb, N * 64, 6, 64);

  // ---- layer 2: 64 -> 128, relu ----
  k_pack_w<64, 128><<<div_up(HID * 2 * HID, TPB), TPB, 0, stream>>>(W2, Wf);
  k_gcn_gemm<64, 128><<<gemm_grid, 32 * gemm_waves_per_blk, 0, stream>>>(Xb, Wf, dinv, G, ACC, mtiles);
  k_scatter<<<div_up(E * (128 / 4), TPB), TPB, 0, stream>>>(src, dst, G, ACC, E, 128, 5);
  k_post_relu_bf16<<<div_up(N * 128, TPB), TPB, 0, stream>>>(ACC, dinv, b2, Xb, N * 128, 7, 128);

  // ---- layer 3: 128 -> 64, no relu ----
  k_pack_w<128, 64><<<div_up(2 * HID * HID, TPB), TPB, 0, stream>>>(W3, Wf);
  k_gcn_gemm<128, 64><<<gemm_grid, 32 * gemm_waves_per_blk, 0, stream>>>(Xb, Wf, dinv, G, ACC, mtiles);
  k_scatter<<<div_up(E * (64 / 4), TPB), TPB, 0, stream>>>(src, dst, G, ACC, E, 64, 4);
  k_post_f32<<<div_up(N * 64, TPB), TPB, 0, stream>>>(ACC, dinv, b3, G, N * 64, 6, 64);  // H -> G buffer

  // ---- global max pool over 512 graphs ----
  k_maxpool<<<512, 64, 0, stream>>>(G, batch, (float*)d_out, N, 64);
}
